// NUFTPolygonEncoder_15461882265815
// MI455X (gfx1250) — matrix operand order repaired and checked by reading, not compiled, not run on hardware
//
#include <hip/hip_runtime.h>
#include <hip/hip_bf16.h>
#include <math.h>

// ---- problem constants (match reference) ----
#define FXN   64
#define FYH   33            // FY//2 + 1
#define NVERT 256
#define BATCH 32
#define NUFT_DIM (FXN * FYH * 2)   // 4224
#define HID   1024
#define OUTD  512

typedef __attribute__((ext_vector_type(16))) __bf16 v16bf;
typedef __attribute__((ext_vector_type(8)))  float  v8f;

// =====================================================================
// Kernel 1: NUFT polygon spectrum -> feats (BATCH x NUFT_DIM, f32)
// One thread per (b, kx, ky) frequency; loop over 256 edges.
// sigma of vertex n+1 is reused as sigma_a of edge n+1 -> one sincos/edge.
// =====================================================================
__global__ __launch_bounds__(256) void nuft_feats_kernel(
    const float* __restrict__ V,      // (B, NV, 2)
    float* __restrict__ feats)        // (B, NUFT_DIM)
{
    __shared__ float sx[NVERT];
    __shared__ float sy[NVERT];

    const int b = blockIdx.x;
    const float* Vb = V + (size_t)b * NVERT * 2;
    for (int i = threadIdx.x; i < NVERT; i += blockDim.x) {
        sx[i] = Vb[2 * i + 0];
        sy[i] = Vb[2 * i + 1];
    }
    __syncthreads();

    const int p = blockIdx.y * blockDim.x + threadIdx.x;  // 0 .. 2111
    if (p >= FXN * FYH) return;

    const int x = p / FYH;
    const int y = p - x * FYH;
    const float kx  = (x < FXN / 2) ? (float)x : (float)(x - FXN);
    const float PI  = 3.14159265358979323846f;
    const float omx = PI * kx;          // 2*pi/TX * kx, TX = 2
    const float omy = PI * (float)y;    // 2*pi/TY * ky, TY = 2

    // sigma / sincos of vertex 0 (also the wrap-around vertex)
    const float sig0 = omx * sx[0] + omy * sy[0];
    float sn0, cs0;
    __sincosf(sig0, &sn0, &cs0);

    float sa = sig0, ca = cs0, sna = sn0;   // state for vertex n
    float accr = 0.0f, acci = 0.0f;

    for (int n = 0; n < NVERT; ++n) {
        const int n1 = (n + 1) & (NVERT - 1);
        float xb, yb, sb, cb, snb;
        if (n1 == 0) {              // wrap: reuse vertex-0 values
            xb = sx[0]; yb = sy[0];
            sb = sig0; cb = cs0; snb = sn0;
        } else {
            xb = sx[n1]; yb = sy[n1];
            sb = omx * xb + omy * yb;
            __sincosf(sb, &snb, &cb);
        }
        const float xa = sx[n], ya = sy[n];
        const float C   = 0.5f * (xa * yb - ya * xb);  // signed tri area
        const float dab = sa - sb;

        const float d1 =  dab * sa;
        const float d2 = -dab * sb;
        const float d3 =  sa * sb;
        const float r1 = (d1 == 0.0f) ? 0.0f : 1.0f / d1;
        const float r2 = (d2 == 0.0f) ? 0.0f : 1.0f / d2;
        const float r3 = (d3 == 0.0f) ? 0.0f : 1.0f / d3;

        // e^{-i s} = cos s - i sin s
        const float Sr =  ca * r1 + cb * r2 + r3;
        const float Si = -sna * r1 - snb * r2;

        accr += C * Sr;
        acci += C * Si;

        sa = sb; ca = cb; sna = snb;       // roll edge state
    }

    const float SCALE = (float)(FXN * FXN) / 4.0f;   // 64*64/(2*2) = 1024
    float Fr = -2.0f * SCALE * accr;
    float Fi = -2.0f * SCALE * acci;
    if (!isfinite(Fr)) Fr = 0.0f;                    // mirrors nan_to_num
    if (!isfinite(Fi)) Fi = 0.0f;

    float* fb = feats + (size_t)b * NUFT_DIM;
    fb[2 * p + 0] = Fr;
    fb[2 * p + 1] = Fi;
}

// =====================================================================
// Kernel 2/3: skinny GEMM, M = 32 fixed, N/K compile-time so the 16
// column-strided weight loads use immediate offsets off one base VGPR
// pair (23*N*4 <= 94208 B fits the signed 24-bit IOFFSET), and the row
// base advances with a single add per K-step.
//
// One wave per 16-column tile; computes BOTH 16-row M-tiles against a
// single shared B fragment (halves weight traffic, which is the roofline
// limiter at M=32). f32 weights/activations are streamed once and
// converted in-register (v_cvt_pk_bf16_f32) for v_wmma_f32_16x16x32_bf16.
//
// A-fragment (16x32 bf16): lane m = lane&15, h = lane>>4 holds
//   K in [8h, 8h+8) and [16+8h, 16+8h+8) of row m.
// B-fragment (32x16 bf16): lane holds K in [8h, 8h+8) and [16+8h, 16+8h+8) of col lane&15.
// C/D: element v -> (M = v + 8h, N = lane&15).
// =====================================================================
__device__ __forceinline__ void cvt8(v16bf& dst, int base, float4 u0, float4 u1) {
    dst[base + 0] = (__bf16)u0.x; dst[base + 1] = (__bf16)u0.y;
    dst[base + 2] = (__bf16)u0.z; dst[base + 3] = (__bf16)u0.w;
    dst[base + 4] = (__bf16)u1.x; dst[base + 5] = (__bf16)u1.y;
    dst[base + 6] = (__bf16)u1.z; dst[base + 7] = (__bf16)u1.w;
}

template <int RELU, int N, int K>
__global__ __launch_bounds__(32) void wmma_gemm_m32(
    const float* __restrict__ A,     // 32 x K, row major f32
    const float* __restrict__ W,     // K x N,  row major f32
    const float* __restrict__ bias,  // N
    float* __restrict__ D)           // 32 x N
{
    const int lane = threadIdx.x & 31;
    const int h    = lane >> 4;
    const int m    = lane & 15;
    const int col  = blockIdx.x * 16 + m;

    const float* A0 = A + m * K;          // rows 0..15
    const float* A1 = A + (16 + m) * K;   // rows 16..31
    const float* Wp = W + (8 * h) * N + col;  // advances by 32*N per step

    v8f acc0 = {};
    v8f acc1 = {};

    for (int k0 = 0; k0 < K; k0 += 32, Wp += 32 * N) {
        v16bf a0, a1, bfrag;

        // A fragments: two contiguous 8-float runs per lane -> b128 pairs
        const float* p0 = A0 + k0 + 8 * h;
        const float* p1 = A0 + k0 + 16 + 8 * h;
        cvt8(a0, 0, *(const float4*)(p0), *(const float4*)(p0 + 4));
        cvt8(a0, 8, *(const float4*)(p1), *(const float4*)(p1 + 4));
        const float* q0 = A1 + k0 + 8 * h;
        const float* q1 = A1 + k0 + 16 + 8 * h;
        cvt8(a1, 0, *(const float4*)(q0), *(const float4*)(q0 + 4));
        cvt8(a1, 8, *(const float4*)(q1), *(const float4*)(q1 + 4));

        // B fragment: 16 row-strided loads, constant immediate offsets,
        // lane-coalesced across the 16 columns of the tile.
#pragma unroll
        for (int e = 0; e < 16; ++e)
            bfrag[e] = (__bf16)Wp[(e < 8 ? e : e + 8) * N];

        if (k0 + 32 < K)  // hint: pull next weight K-block toward L2
            __builtin_prefetch(Wp + 32 * N, 0, 1);

        acc0 = __builtin_amdgcn_wmma_f32_16x16x32_bf16(
                   false, a0, false, bfrag, (short)0, acc0, false, false);
        acc1 = __builtin_amdgcn_wmma_f32_16x16x32_bf16(
                   false, a1, false, bfrag, (short)0, acc1, false, false);
    }

    const float bv = bias[col];
#pragma unroll
    for (int v = 0; v < 8; ++v) {
        const int mo = v + 8 * h;
        float x0 = acc0[v] + bv;
        float x1 = acc1[v] + bv;
        if (RELU) { x0 = fmaxf(x0, 0.0f); x1 = fmaxf(x1, 0.0f); }
        D[mo * N + col]        = x0;
        D[(16 + mo) * N + col] = x1;
    }
}

// =====================================================================
// Launch: feats -> GEMM1(+bias,ReLU) -> GEMM2(+bias)
// =====================================================================
extern "C" void kernel_launch(void* const* d_in, const int* in_sizes, int n_in,
                              void* d_out, int out_size, void* d_ws, size_t ws_size,
                              hipStream_t stream) {
    (void)in_sizes; (void)n_in; (void)out_size; (void)ws_size;

    const float* V  = (const float*)d_in[0];   // (32, 256, 2)
    const float* W1 = (const float*)d_in[1];   // (4224, 1024)
    const float* b1 = (const float*)d_in[2];   // (1024,)
    const float* W2 = (const float*)d_in[3];   // (1024, 512)
    const float* b2 = (const float*)d_in[4];   // (512,)
    float* out = (float*)d_out;                // (32, 512)

    float* feats = (float*)d_ws;                       // 32*4224 f32
    float* hbuf  = feats + (size_t)BATCH * NUFT_DIM;   // 32*1024 f32

    // 1) spectrum features: 32 batches x 9 chunks of 256 freq points
    nuft_feats_kernel<<<dim3(BATCH, (FXN * FYH + 255) / 256), 256, 0, stream>>>(V, feats);

    // 2) feats @ W1 + b1, ReLU  (K = 4224, N = 1024)
    wmma_gemm_m32<1, HID, NUFT_DIM><<<HID / 16, 32, 0, stream>>>(feats, W1, b1, hbuf);

    // 3) h @ W2 + b2            (K = 1024, N = 512)
    wmma_gemm_m32<0, OUTD, HID><<<OUTD / 16, 32, 0, stream>>>(hbuf, W2, b2, out);
}
